// GATv2Backbone_81011673137798
// MI455X (gfx1250) — matrix operand order repaired and checked by reading, not compiled, not run on hardware
//
#include <hip/hip_runtime.h>

#define BB 128
#define TT 64
#define NN9 9
#define NUM_NODES (BB*TT*NN9)   // 73728
#define HID 256
#define D_ENC 128
#define INV_REAL_EDGES (1.0f/203648.0f)

typedef __bf16 bf_t;
typedef bf_t v16bf __attribute__((ext_vector_type(16)));
typedef bf_t v8bf  __attribute__((ext_vector_type(8)));
typedef float v8f  __attribute__((ext_vector_type(8)));

// ---- bf16 conversion -------------------------------------------------------
__global__ void cvt_f32_to_bf16(const float* __restrict__ in, bf_t* __restrict__ out, int n) {
  int i = blockIdx.x * blockDim.x + threadIdx.x;
  if (i < n) out[i] = (bf_t)in[i];
}

// ---- encoder: x[v,d] = sum_f feats[v,f]*enc_W[n,d,f] + enc_b[n,d] ---------
__global__ void encoder_kernel(const float* __restrict__ nf, const float* __restrict__ encW,
                               const float* __restrict__ encb, float* __restrict__ x) {
  int idx = blockIdx.x * blockDim.x + threadIdx.x;   // v*128 + d
  if (idx >= NUM_NODES * D_ENC) return;
  int v = idx >> 7, d = idx & 127;
  int n = v % NN9;
  const float* w = encW + ((size_t)n * D_ENC + d) * 14;
  const float* f = nf + (size_t)v * 10;
  float acc = encb[n * D_ENC + d];
#pragma unroll
  for (int j = 0; j < 10; ++j) acc += f[j] * w[j];
  float o0 = (n < 7) ? 1.f : 0.f;
  float o1 = (n == 7) ? 1.f : 0.f;
  float o2 = (n == 8) ? 1.f : 0.f;
  float dist = fabsf((float)(n - 7));
  acc += o0 * w[10] + o1 * w[11] + o2 * w[12] + dist * w[13];
  x[idx] = acc;
}

// ---- fill = mean over real edges of (pos[dst]-pos[src]) --------------------
// spatial pairs cancel; temporal edges telescope: sum_{b,n}(pos[b,T-1,n]-pos[b,0,n])
__global__ void fill_kernel(const float* __restrict__ nf, float* __restrict__ fill) {
  __shared__ float sm[3 * 256];
  int tid = threadIdx.x;
  float s0 = 0.f, s1 = 0.f, s2 = 0.f;
  for (int i = tid; i < BB * NN9; i += 256) {
    int b = i / NN9, n = i % NN9;
    size_t vTop = ((size_t)b * TT + (TT - 1)) * NN9 + n;
    size_t vBot = ((size_t)b * TT) * NN9 + n;
    const float* pt = nf + vTop * 10;
    const float* pb = nf + vBot * 10;
    s0 += pt[0] - pb[0]; s1 += pt[1] - pb[1]; s2 += pt[2] - pb[2];
  }
  sm[tid] = s0; sm[256 + tid] = s1; sm[512 + tid] = s2;
  __syncthreads();
  for (int off = 128; off > 0; off >>= 1) {
    if (tid < off) {
      sm[tid] += sm[tid + off];
      sm[256 + tid] += sm[256 + tid + off];
      sm[512 + tid] += sm[512 + tid + off];
    }
    __syncthreads();
  }
  if (tid == 0) {
    fill[0] = sm[0] * INV_REAL_EDGES;
    fill[1] = sm[256] * INV_REAL_EDGES;
    fill[2] = sm[512] * INV_REAL_EDGES;
  }
}

// ---- WMMA GEMM: Y(M x 256) = X(M x K) @ W(256 x K)^T + bias ----------------
// A k order chosen here (not the ISA table order): lanes 0-15 hold K=0..15 contiguous;
// lanes 16-31: K=16..31.   B uses the same order: lanes 0-15 K=0..15 contiguous,
// lanes 16-31 K=16..31 (W rows are contiguous in K, so a straight 32B load).
__device__ __forceinline__ v16bf load_a16(const bf_t* p) {
  v8bf lo = *(const v8bf*)p;
  v8bf hi = *(const v8bf*)(p + 8);
  return __builtin_shufflevector(lo, hi, 0,1,2,3,4,5,6,7,8,9,10,11,12,13,14,15);
}

__global__ __launch_bounds__(256) void gemm_bf16_wmma(
    const bf_t* __restrict__ X, const bf_t* __restrict__ W,
    const float* __restrict__ bias, float* __restrict__ Y, int K) {
  const int lane = threadIdx.x & 31;
  const int wave = threadIdx.x >> 5;
  const int m0 = (blockIdx.x * 2 + (wave >> 2)) * 16;
  const int n0 = (wave & 3) * 64;
  const int half = lane >> 4;
  const int idx = lane & 15;
  const int abase = half * 16;
  const int bbase = half * 16;

  const bf_t* xrow = X + (size_t)(m0 + idx) * K;
  const bf_t* w0 = W + (size_t)(n0 + idx) * K;
  const bf_t* w1 = w0 + (size_t)16 * K;
  const bf_t* w2 = w0 + (size_t)32 * K;
  const bf_t* w3 = w0 + (size_t)48 * K;

  v8f acc0 = {}, acc1 = {}, acc2 = {}, acc3 = {};
  for (int k = 0; k < K; k += 32) {
    v16bf a  = load_a16(xrow + k + abase);
    v16bf b0 = *(const v16bf*)(w0 + k + bbase);
    v16bf b1 = *(const v16bf*)(w1 + k + bbase);
    v16bf b2 = *(const v16bf*)(w2 + k + bbase);
    v16bf b3 = *(const v16bf*)(w3 + k + bbase);
    acc0 = __builtin_amdgcn_wmma_f32_16x16x32_bf16(false, a, false, b0, (short)0, acc0, false, false);
    acc1 = __builtin_amdgcn_wmma_f32_16x16x32_bf16(false, a, false, b1, (short)0, acc1, false, false);
    acc2 = __builtin_amdgcn_wmma_f32_16x16x32_bf16(false, a, false, b2, (short)0, acc2, false, false);
    acc3 = __builtin_amdgcn_wmma_f32_16x16x32_bf16(false, a, false, b3, (short)0, acc3, false, false);
  }
  const int rbase = m0 + half * 8;
  const int c = n0 + idx;
  float bc0 = bias[c], bc1 = bias[c + 16], bc2 = bias[c + 32], bc3 = bias[c + 48];
#pragma unroll
  for (int r = 0; r < 8; ++r) {
    float* yr = Y + (size_t)(rbase + r) * HID;
    yr[c]      = acc0[r] + bc0;
    yr[c + 16] = acc1[r] + bc1;
    yr[c + 32] = acc2[r] + bc2;
    yr[c + 48] = acc3[r] + bc3;
  }
}

// ---- fused GAT attention + residual + LayerNorm + SiLU ---------------------
// one wave per node; lane owns channels [lane*8, lane*8+8) -> head = lane/8.
__global__ __launch_bounds__(256) void gat_attn_kernel(
    const float* __restrict__ XL, const float* __restrict__ XR, const float* __restrict__ XS,
    const float* __restrict__ nf, const float* __restrict__ fill,
    const float* __restrict__ att, const float* __restrict__ eW,
    const float* __restrict__ gg, const float* __restrict__ bbv,
    float* __restrict__ Xout) {
  const int wave = threadIdx.x >> 5;
  const int lane = threadIdx.x & 31;
  const int v = blockIdx.x * 8 + wave;
  if (v >= NUM_NODES) return;
  const int n = v % NN9;
  const int t = (v / NN9) % TT;
  const int c0 = lane * 8;

  float xr[8], xs[8], ew0[8], ew1[8], ew2[8], attv[8];
#pragma unroll
  for (int j = 0; j < 8; ++j) {
    int c = c0 + j;
    xr[j] = XR[(size_t)v * HID + c];
    xs[j] = XS[(size_t)v * HID + c];
    ew0[j] = eW[c * 3 + 0]; ew1[j] = eW[c * 3 + 1]; ew2[j] = eW[c * 3 + 2];
    attv[j] = att[c];
  }
  const float px = nf[(size_t)v * 10 + 0];
  const float py = nf[(size_t)v * 10 + 1];
  const float pz = nf[(size_t)v * 10 + 2];

  int u[4]; bool valid[4]; float ea0[4], ea1[4], ea2[4];
  valid[0] = (n > 0);  u[0] = valid[0] ? v - 1 : v;
  valid[1] = (n < 8);  u[1] = valid[1] ? v + 1 : v;
  valid[2] = (t > 0);  u[2] = valid[2] ? v - NN9 : v;
  valid[3] = true;     u[3] = v;
#pragma unroll
  for (int e = 0; e < 3; ++e) {
    const float* pu = nf + (size_t)u[e] * 10;
    ea0[e] = px - pu[0]; ea1[e] = py - pu[1]; ea2[e] = pz - pu[2];
  }
  ea0[3] = fill[0]; ea1[3] = fill[1]; ea2[3] = fill[2];

  float xlu[4][8], lg[4];
#pragma unroll
  for (int e = 0; e < 4; ++e) {
    const float* xlrow = XL + (size_t)u[e] * HID;
    float p = 0.f;
#pragma unroll
    for (int j = 0; j < 8; ++j) {
      float xl = xlrow[c0 + j];
      xlu[e][j] = xl;
      float m = xl + xr[j] + ea0[e] * ew0[j] + ea1[e] * ew1[j] + ea2[e] * ew2[j];
      m = (m > 0.f) ? m : 0.2f * m;
      p += m * attv[j];
    }
    p += __shfl_xor(p, 1, 32);   // reduce within the 8-lane head group
    p += __shfl_xor(p, 2, 32);
    p += __shfl_xor(p, 4, 32);
    lg[e] = valid[e] ? p : -1e30f;
  }
  float mx = fmaxf(fmaxf(lg[0], lg[1]), fmaxf(lg[2], lg[3]));
  float wgt[4], dsum = 0.f;
#pragma unroll
  for (int e = 0; e < 4; ++e) { wgt[e] = expf(lg[e] - mx); dsum += wgt[e]; }
  float inv = 1.f / dsum;

  float out[8];
#pragma unroll
  for (int j = 0; j < 8; ++j) out[j] = xs[j];
#pragma unroll
  for (int e = 0; e < 4; ++e) {
    float w = wgt[e] * inv;
#pragma unroll
    for (int j = 0; j < 8; ++j) out[j] += w * xlu[e][j];
  }
  // LayerNorm over 256 channels (full-wave reduction)
  float s = 0.f;
#pragma unroll
  for (int j = 0; j < 8; ++j) s += out[j];
  s += __shfl_xor(s, 1, 32); s += __shfl_xor(s, 2, 32); s += __shfl_xor(s, 4, 32);
  s += __shfl_xor(s, 8, 32); s += __shfl_xor(s, 16, 32);
  float mu = s * (1.f / 256.f);
  float vs = 0.f;
#pragma unroll
  for (int j = 0; j < 8; ++j) { float d = out[j] - mu; vs += d * d; }
  vs += __shfl_xor(vs, 1, 32); vs += __shfl_xor(vs, 2, 32); vs += __shfl_xor(vs, 4, 32);
  vs += __shfl_xor(vs, 8, 32); vs += __shfl_xor(vs, 16, 32);
  float rstd = rsqrtf(vs * (1.f / 256.f) + 1e-5f);
#pragma unroll
  for (int j = 0; j < 8; ++j) {
    int c = c0 + j;
    float xn = (out[j] - mu) * rstd * gg[c] + bbv[c];
    float sg = 1.f / (1.f + expf(-xn));
    Xout[(size_t)v * HID + c] = xn * sg;
  }
}

// ---- final mean over T*N per batch -----------------------------------------
__global__ void mean_kernel(const float* __restrict__ X, float* __restrict__ out) {
  int b = blockIdx.x, c = threadIdx.x;
  const float* base = X + (size_t)b * (TT * NN9) * HID + c;
  float s = 0.f;
  for (int i = 0; i < TT * NN9; ++i) s += base[(size_t)i * HID];
  out[b * HID + c] = s * (1.f / (TT * NN9));
}

// ---------------------------------------------------------------------------
extern "C" void kernel_launch(void* const* d_in, const int* in_sizes, int n_in,
                              void* d_out, int out_size, void* d_ws, size_t ws_size,
                              hipStream_t stream) {
  (void)in_sizes; (void)n_in; (void)out_size; (void)ws_size;
  const float* node_feats = (const float*)d_in[0];
  const float* enc_W = (const float*)d_in[1];
  const float* enc_b = (const float*)d_in[2];

  const size_t NBUF = (size_t)NUM_NODES * HID;          // 18,874,368 floats
  char* ws = (char*)d_ws;
  float* x   = (float*)ws;                        ws += NBUF * sizeof(float);
  float* XL  = (float*)ws;                        ws += NBUF * sizeof(float);
  float* XR  = (float*)ws;                        ws += NBUF * sizeof(float);
  float* XS  = (float*)ws;                        ws += NBUF * sizeof(float);
  bf_t*  xh  = (bf_t*)ws;                         ws += NBUF * sizeof(bf_t);
  bf_t*  wh  = (bf_t*)ws;                         ws += (size_t)HID * HID * sizeof(bf_t);
  float* fill = (float*)ws;                       ws += 256;

  // encoder + self-loop eattr fill
  {
    int n = NUM_NODES * D_ENC;
    encoder_kernel<<<(n + 255) / 256, 256, 0, stream>>>(node_feats, enc_W, enc_b, x);
    fill_kernel<<<1, 256, 0, stream>>>(node_feats, fill);
  }

  int din = D_ENC;
  for (int L = 0; L < 3; ++L) {
    const float* lW   = (const float*)d_in[3 + 10 * L + 0];
    const float* lb   = (const float*)d_in[3 + 10 * L + 1];
    const float* rW   = (const float*)d_in[3 + 10 * L + 2];
    const float* rb   = (const float*)d_in[3 + 10 * L + 3];
    const float* attp = (const float*)d_in[3 + 10 * L + 4];
    const float* eWp  = (const float*)d_in[3 + 10 * L + 5];
    const float* sW   = (const float*)d_in[3 + 10 * L + 6];
    const float* bias = (const float*)d_in[3 + 10 * L + 7];
    const float* gg   = (const float*)d_in[3 + 10 * L + 8];
    const float* bbv  = (const float*)d_in[3 + 10 * L + 9];

    const int xcnt = NUM_NODES * din;
    const int wcnt = HID * din;
    const int gemm_grid = NUM_NODES / 32;   // 2 row-tiles of 16 per block

    cvt_f32_to_bf16<<<(xcnt + 255) / 256, 256, 0, stream>>>(x, xh, xcnt);

    cvt_f32_to_bf16<<<(wcnt + 255) / 256, 256, 0, stream>>>(lW, wh, wcnt);
    gemm_bf16_wmma<<<gemm_grid, 256, 0, stream>>>(xh, wh, lb, XL, din);

    cvt_f32_to_bf16<<<(wcnt + 255) / 256, 256, 0, stream>>>(rW, wh, wcnt);
    gemm_bf16_wmma<<<gemm_grid, 256, 0, stream>>>(xh, wh, rb, XR, din);

    cvt_f32_to_bf16<<<(wcnt + 255) / 256, 256, 0, stream>>>(sW, wh, wcnt);
    gemm_bf16_wmma<<<gemm_grid, 256, 0, stream>>>(xh, wh, bias, XS, din);

    gat_attn_kernel<<<NUM_NODES / 8, 256, 0, stream>>>(XL, XR, XS, node_feats, fill,
                                                       attp, eWp, gg, bbv, x);
    din = HID;
  }

  mean_kernel<<<BB, HID, 0, stream>>>(x, (float*)d_out);
}
